// VARQuantizer_7146825580549
// MI455X (gfx1250) — matrix operand order repaired and checked by reading, compiled, not run on hardware
//
#include <hip/hip_runtime.h>
#include <hip/hip_bf16.h>

typedef __bf16 bf16_t;
typedef __attribute__((ext_vector_type(16))) __bf16 v16bf;
typedef __attribute__((ext_vector_type(8)))  __bf16 v8bf;
typedef __attribute__((ext_vector_type(8)))  float  v8f;

#define BATCH 4
#define TLEN  8192
#define CBIG  512
#define CSM   16

__device__ __forceinline__ float snake_f(float x, float a) {
    float s = __sinf(a * x);
    return x + s * s / (a + 1e-9f);
}

__device__ __forceinline__ v8bf vzero8() {
    v8bf z;
#pragma unroll
    for (int j = 0; j < 8; ++j) z[j] = (bf16_t)0.0f;
    return z;
}

// ---------------------------------------------------------------------------
// Weight pre-swizzle: w[Cout][Cin][ks] f32 -> Wf[chunk][co][32] bf16,
// flatK = k*Cin + ci, chunk = flatK/32, zero-padded past ks*Cin.
// Uses contiguous k order (B fragments match it): lane m=l&15 reads its row's
// chunk-local offsets [kb,kb+16), kb=16*(l>>4).
// ---------------------------------------------------------------------------
__global__ void prep_weights(const float* __restrict__ w, bf16_t* __restrict__ wf,
                             int Cin, int Cout, int ks, int nChunks) {
    int idx = blockIdx.x * blockDim.x + threadIdx.x;
    int total = nChunks * Cout * 32;
    if (idx >= total) return;
    int kl = idx & 31;
    int co = (idx >> 5) % Cout;
    int c  = idx / (32 * Cout);
    int flatK = c * 32 + kl;
    float val = 0.0f;
    if (flatK < ks * Cin) {
        int k  = flatK / Cin;
        int ci = flatK - k * Cin;
        val = w[((size_t)co * Cin + ci) * ks + k];
    }
    wf[((size_t)c * Cout + co) * 32 + kl] = (bf16_t)val;
}

// x f32 [B][C][T] -> out bf16 [B][T][C] with snake applied
__global__ void snake_ct_to_tc(const float* __restrict__ x, const float* __restrict__ alpha,
                               bf16_t* __restrict__ out, int C, int T) {
    size_t idx = (size_t)blockIdx.x * blockDim.x + threadIdx.x;
    size_t total = (size_t)BATCH * T * C;
    if (idx >= total) return;
    int c = (int)(idx % C);
    size_t bt = idx / C;
    int t = (int)(bt % T);
    int b = (int)(bt / T);
    float v = x[((size_t)b * C + c) * T + t];
    out[idx] = (bf16_t)snake_f(v, alpha[c]);
}

__global__ void cvt_f32_bf16(const float* __restrict__ in, bf16_t* __restrict__ out, size_t n) {
    size_t i = (size_t)blockIdx.x * blockDim.x + threadIdx.x;
    if (i < n) out[i] = (bf16_t)in[i];
}

// up = repeat(r, scale) along T, then snake: out bf16 [B][T][16]
__global__ void upsample_snake(const float* __restrict__ r, const float* __restrict__ alpha,
                               int scale, bf16_t* __restrict__ out) {
    size_t idx = (size_t)blockIdx.x * blockDim.x + threadIdx.x;
    size_t total = (size_t)BATCH * TLEN * CSM;
    if (idx >= total) return;
    int c = (int)(idx & 15);
    size_t bt = idx >> 4;
    int t = (int)(bt % TLEN);
    int b = (int)(bt / TLEN);
    float v = r[((size_t)b * (TLEN / scale) + t / scale) * CSM + c];
    out[idx] = (bf16_t)snake_f(v, alpha[c]);
}

// ---------------------------------------------------------------------------
// FSQ at subsampled points
// ---------------------------------------------------------------------------
__global__ void fsq_kernel(const float* __restrict__ z,
                           const float* __restrict__ win, const float* __restrict__ bin_,
                           const float* __restrict__ wout, const float* __restrict__ bout,
                           int scale, int P, float* __restrict__ r, int* __restrict__ codes) {
    int idx = blockIdx.x * blockDim.x + threadIdx.x;
    if (idx >= BATCH * P) return;
    int b = idx / P, p = idx - b * P;
    int t = p * scale;
    const float* zl = z + ((size_t)b * TLEN + t) * CSM;
    float zv[16];
#pragma unroll
    for (int c = 0; c < 16; ++c) zv[c] = zl[c];

    const float levels[5] = {8.f, 8.f, 8.f, 8.f, 5.f};
    const float hw[5]     = {4.f, 4.f, 4.f, 4.f, 2.f};
    const float basis[5]  = {1.f, 8.f, 64.f, 512.f, 4096.f};
    float q[5];
    float fidx = 0.f;
#pragma unroll
    for (int j = 0; j < 5; ++j) {
        float half_l = (levels[j] - 1.f) * 1.001f * 0.5f;
        float offset = (j < 4) ? 0.5f : 0.0f;
        float shift  = (j < 4) ? atanhf(offset / half_l) : 0.0f;
        float s = bin_[j];
        for (int c = 0; c < 16; ++c) s += win[j * 16 + c] * zv[c];
        float bounded = tanhf(s + shift) * half_l - offset;
        q[j] = rintf(bounded);
        fidx += (q[j] + hw[j]) * basis[j];
    }
    codes[idx] = (int)fidx;
    float* rr = r + (size_t)idx * CSM;
#pragma unroll
    for (int c = 0; c < 16; ++c) {
        float o = bout[c];
        for (int j = 0; j < 5; ++j) o += wout[c * 5 + j] * (q[j] / hw[j]);
        rr[c] = o;
    }
}

// ---------------------------------------------------------------------------
// Implicit-GEMM conv1d with v_wmma_f32_16x16x32_bf16, LDS-staged B tile.
// S  : bf16 activations [B][T][Cin]   (Cin power of two; lc = log2 Cin)
// Wf : fragment-native weights [chunk][Cout][32] (L2-resident)
// Block: coWaves waves, each wave owns 16*msCount co x 32 t.
// Activation rows [tBase-pad, tBase-pad+nRows) staged once into LDS
// (stride Cin+8 halves -> conflict-free ds_load_b128), boundary zeros
// written at stage time, so the K loop is branch-free: 4 WMMA per chunk.
// Epilogue modes:
//  0: outTC = bf16(snake(acc+bias, alpha2))             [B][T][Cout]
//  1: v=acc+bias(+residCT[c][t]); outCT f32 [B][C][T] and/or outTC bf16 [B][T][C]
//  2: v=acc+bias+Rup[t/scale]; Z-=v; ZQ+=v  (f32 [B][T][16])
//  3: outF32TC = acc+bias                               [B][T][Cout] f32
//  4: v=acc+bias; outCT[c][t]=v; outTC=bf16(snake(v,alpha2))
// ---------------------------------------------------------------------------
__global__ void conv_wmma(const bf16_t* __restrict__ S, const bf16_t* __restrict__ Wf,
                          const float* __restrict__ bias,
                          int T, int Cin, int lc, int Cout, int ks, int pad,
                          int nChunks, int msCount, int strideH, int nRows, int mode,
                          const float* __restrict__ alpha2,
                          bf16_t* __restrict__ outTC,
                          const float* __restrict__ residCT,
                          float* __restrict__ outCT,
                          float* __restrict__ outF32TC,
                          float* __restrict__ Z, float* __restrict__ ZQ,
                          const float* __restrict__ Rup, int scale) {
    extern __shared__ __align__(16) char smem_raw[];
    bf16_t* sm = (bf16_t*)smem_raw;

    const int lane   = threadIdx.x & 31;
    const int wave   = threadIdx.x >> 5;
    const int laneLo = lane & 15;
    const int laneHi = lane >> 4;
    const int b      = blockIdx.z;
    const int coWaves = blockDim.x >> 5;
    const int coBase = (blockIdx.y * coWaves + wave) * (16 * msCount);
    const int tBase  = blockIdx.x * 32;

    const bf16_t* Sb = S + (size_t)b * T * Cin;

    // ---- stage activation tile rows into LDS (16B vector groups)
    {
        const int grpPerRow = 1 << (lc - 3);
        const int totalGrp  = nRows * grpPerRow;
        for (int g = threadIdx.x; g < totalGrp; g += blockDim.x) {
            int row = g >> (lc - 3);
            int grp = g & (grpPerRow - 1);
            int t   = tBase - pad + row;
            v8bf v  = vzero8();
            if ((unsigned)t < (unsigned)T)
                v = *(const v8bf*)(Sb + ((size_t)t << lc) + (grp << 3));
            *(v8bf*)(sm + row * strideH + (grp << 3)) = v;
        }
    }
    __syncthreads();

    v8f acc00 = {0.f,0.f,0.f,0.f,0.f,0.f,0.f,0.f};
    v8f acc01 = acc00, acc10 = acc00, acc11 = acc00;

    const int kb   = laneHi * 16;
    const int cmsk = Cin - 1;
    for (int c = 0; c < nChunks; ++c) {
        // ---- A fragments (fragment-native weights, straight 16B global loads)
        const bf16_t* arow0 = Wf + ((size_t)c * Cout + (coBase + laneLo)) * 32;
        v16bf a0, a1;
        ((v8bf*)&a0)[0] = *(const v8bf*)(arow0 + kb);
        ((v8bf*)&a0)[1] = *(const v8bf*)(arow0 + kb + 8);
        if (msCount == 2) {
            const bf16_t* arow1 = arow0 + 16 * 32;
            ((v8bf*)&a1)[0] = *(const v8bf*)(arow1 + kb);
            ((v8bf*)&a1)[1] = *(const v8bf*)(arow1 + kb + 8);
        }
        if (c + 1 < nChunks)
            __builtin_prefetch(arow0 + (size_t)Cout * 32, 0, 0);

        // ---- B fragments from LDS (tap/ci via shift/mask; no branches)
        const int Kstart = c * 32 + laneHi * 16;
        const int tap    = Kstart >> lc;
        const int ciB    = Kstart & cmsk;
        const int r0     = laneLo + tap;
        const bf16_t* p0 = sm + r0 * strideH + ciB;
        const bf16_t* p1 = p0 + 16 * strideH;
        v16bf b0, b1;
        ((v8bf*)&b0)[0] = *(const v8bf*)(p0);
        ((v8bf*)&b0)[1] = *(const v8bf*)(p0 + 8);
        ((v8bf*)&b1)[0] = *(const v8bf*)(p1);
        ((v8bf*)&b1)[1] = *(const v8bf*)(p1 + 8);

        acc00 = __builtin_amdgcn_wmma_f32_16x16x32_bf16(false, a0, false, b0,
                                                        (short)0, acc00, false, false);
        acc01 = __builtin_amdgcn_wmma_f32_16x16x32_bf16(false, a0, false, b1,
                                                        (short)0, acc01, false, false);
        if (msCount == 2) {
            acc10 = __builtin_amdgcn_wmma_f32_16x16x32_bf16(false, a1, false, b0,
                                                            (short)0, acc10, false, false);
            acc11 = __builtin_amdgcn_wmma_f32_16x16x32_bf16(false, a1, false, b1,
                                                            (short)0, acc11, false, false);
        }
    }

    // ---- Epilogue. D layout: lane n=laneLo (t), m = 8*laneHi + v
    const int mBase = laneHi * 8;
#pragma unroll
    for (int ms = 0; ms < 2; ++ms) {
        if (ms == 1 && msCount != 2) break;
        const int co0 = coBase + ms * 16;
#pragma unroll
        for (int sub = 0; sub < 2; ++sub) {
            v8f acc = (ms == 0) ? (sub ? acc01 : acc00) : (sub ? acc11 : acc10);
            const int t = tBase + sub * 16 + laneLo;
            if (mode == 0) {
                v8bf pk;
#pragma unroll
                for (int v = 0; v < 8; ++v) {
                    int co = co0 + mBase + v;
                    float f = acc[v] + bias[co];
                    pk[v] = (bf16_t)snake_f(f, alpha2[co]);
                }
                *(v8bf*)(outTC + ((size_t)b * T + t) * Cout + co0 + mBase) = pk;
            } else if (mode == 1) {
#pragma unroll
                for (int v = 0; v < 8; ++v) {
                    int co = co0 + mBase + v;
                    float f = acc[v] + bias[co];
                    if (residCT) f += residCT[((size_t)b * Cout + co) * T + t];
                    if (outCT) outCT[((size_t)b * Cout + co) * T + t] = f;
                    if (outTC) outTC[((size_t)b * T + t) * Cout + co] = (bf16_t)f;
                }
            } else if (mode == 2) {
                size_t rbase = ((size_t)b * (T / scale) + t / scale) * CSM;
                size_t zbase = ((size_t)b * T + t) * CSM;
#pragma unroll
                for (int v = 0; v < 8; ++v) {
                    int cc = mBase + v;
                    float f = acc[v] + bias[cc] + Rup[rbase + cc];
                    Z[zbase + cc]  -= f;
                    ZQ[zbase + cc] += f;
                }
            } else if (mode == 3) {
#pragma unroll
                for (int v = 0; v < 8; ++v) {
                    int co = co0 + mBase + v;
                    outF32TC[((size_t)b * T + t) * Cout + co] = acc[v] + bias[co];
                }
            } else { // mode 4
#pragma unroll
                for (int v = 0; v < 8; ++v) {
                    int co = co0 + mBase + v;
                    float f = acc[v] + bias[co];
                    outCT[((size_t)b * Cout + co) * T + t] = f;
                    outTC[((size_t)b * T + t) * Cout + co] = (bf16_t)snake_f(f, alpha2[co]);
                }
            }
        }
    }
}

// ---------------------------------------------------------------------------
static void launch_conv(const bf16_t* S, const bf16_t* Wf, const float* bias,
                        int Cin, int Cout, int ks, int pad, int mode,
                        const float* alpha2, bf16_t* outTC, const float* residCT,
                        float* outCT, float* outF32TC,
                        float* Z, float* ZQ, const float* Rup, int scale,
                        hipStream_t stream) {
    int lc       = __builtin_ctz((unsigned)Cin);
    int nChunks  = (ks * Cin + 31) / 32;
    int msCount  = (Cout >= 32) ? 2 : 1;
    int coWaves  = Cout / (16 * msCount);
    if (coWaves > 8) coWaves = 8;
    int strideH  = Cin + 8;
    int nRows    = 32 + ks;
    size_t shBytes = (size_t)nRows * strideH * 2;
    dim3 grid(TLEN / 32, Cout / (16 * msCount * coWaves), BATCH);
    dim3 block(coWaves * 32);
    conv_wmma<<<grid, block, shBytes, stream>>>(S, Wf, bias, TLEN, Cin, lc, Cout, ks, pad,
                                                nChunks, msCount, strideH, nRows, mode,
                                                alpha2, outTC, residCT, outCT, outF32TC,
                                                Z, ZQ, Rup, scale);
}

extern "C" void kernel_launch(void* const* d_in, const int* in_sizes, int n_in,
                              void* d_out, int out_size, void* d_ws, size_t ws_size,
                              hipStream_t stream) {
    (void)in_sizes; (void)n_in; (void)out_size; (void)ws_size;
    const float* zin       = (const float*)d_in[0];
    const float* inru_a1   = (const float*)d_in[1];
    const float* inru_w1   = (const float*)d_in[2];
    const float* inru_b1   = (const float*)d_in[3];
    const float* inru_a2   = (const float*)d_in[4];
    const float* inru_w2   = (const float*)d_in[5];
    const float* inru_b2   = (const float*)d_in[6];
    const float* in_w      = (const float*)d_in[7];
    const float* in_b      = (const float*)d_in[8];
    const float* fsq_win   = (const float*)d_in[9];
    const float* fsq_bin   = (const float*)d_in[10];
    const float* fsq_wout  = (const float*)d_in[11];
    const float* fsq_bout  = (const float*)d_in[12];
    const float* phi_a1    = (const float*)d_in[13];
    const float* phi_w1    = (const float*)d_in[14];
    const float* phi_b1    = (const float*)d_in[15];
    const float* phi_a2    = (const float*)d_in[16];
    const float* phi_w2    = (const float*)d_in[17];
    const float* phi_b2    = (const float*)d_in[18];
    const float* out_w     = (const float*)d_in[19];
    const float* out_b     = (const float*)d_in[20];
    const float* outru_a1  = (const float*)d_in[21];
    const float* outru_w1  = (const float*)d_in[22];
    const float* outru_b1  = (const float*)d_in[23];
    const float* outru_a2  = (const float*)d_in[24];
    const float* outru_w2  = (const float*)d_in[25];
    const float* outru_b2  = (const float*)d_in[26];

    // ---- workspace layout
    char* ws = (char*)d_ws;
    size_t o = 0;
    bf16_t* sA    = (bf16_t*)(ws + o); o += (size_t)BATCH * TLEN * CBIG * 2;
    bf16_t* sB    = (bf16_t*)(ws + o); o += (size_t)BATCH * TLEN * CBIG * 2;
    float*  zbuf  = (float*) (ws + o); o += (size_t)BATCH * TLEN * CSM * 4;
    float*  zqbuf = (float*) (ws + o); o += (size_t)BATCH * TLEN * CSM * 4;
    float*  rbuf  = (float*) (ws + o); o += (size_t)BATCH * (TLEN / 2) * CSM * 4;
    bf16_t* s16a  = (bf16_t*)(ws + o); o += (size_t)BATCH * TLEN * CSM * 2;
    bf16_t* s16b  = (bf16_t*)(ws + o); o += (size_t)BATCH * TLEN * CSM * 2;
    bf16_t* zqbf  = (bf16_t*)(ws + o); o += (size_t)BATCH * TLEN * CSM * 2;
    bf16_t* w1fi  = (bf16_t*)(ws + o); o += (size_t)112 * CBIG * 32 * 2;
    bf16_t* w2fi  = (bf16_t*)(ws + o); o += (size_t)16 * CBIG * 32 * 2;
    bf16_t* inwf  = (bf16_t*)(ws + o); o += (size_t)16 * CSM * 32 * 2;
    bf16_t* phi1f = (bf16_t*)(ws + o); o += (size_t)4 * CSM * 32 * 2;
    bf16_t* phi2f = (bf16_t*)(ws + o); o += (size_t)1 * CSM * 32 * 2;
    bf16_t* outwf = (bf16_t*)(ws + o); o += (size_t)1 * CBIG * 32 * 2;
    bf16_t* w1fo  = (bf16_t*)(ws + o); o += (size_t)112 * CBIG * 32 * 2;
    bf16_t* w2fo  = (bf16_t*)(ws + o); o += (size_t)16 * CBIG * 32 * 2;

    auto prep = [&](const float* w, bf16_t* wf, int Cin, int Cout, int ks) {
        int nC = (ks * Cin + 31) / 32;
        int total = nC * Cout * 32;
        prep_weights<<<(total + 255) / 256, 256, 0, stream>>>(w, wf, Cin, Cout, ks, nC);
    };
    prep(inru_w1,  w1fi,  CBIG, CBIG, 7);
    prep(inru_w2,  w2fi,  CBIG, CBIG, 1);
    prep(in_w,     inwf,  CBIG, CSM,  1);
    prep(phi_w1,   phi1f, CSM,  CSM,  7);
    prep(phi_w2,   phi2f, CSM,  CSM,  1);
    prep(out_w,    outwf, CSM,  CBIG, 1);
    prep(outru_w1, w1fo,  CBIG, CBIG, 7);
    prep(outru_w2, w2fo,  CBIG, CBIG, 1);

    const size_t nBig = (size_t)BATCH * TLEN * CBIG;
    const size_t nSm  = (size_t)BATCH * TLEN * CSM;

    // ---- in_ru: x + conv1(snake(conv7(snake(x))))
    snake_ct_to_tc<<<(int)((nBig + 255) / 256), 256, 0, stream>>>(zin, inru_a1, sA, CBIG, TLEN);
    launch_conv(sA, w1fi, inru_b1, CBIG, CBIG, 7, 3, /*mode*/0, inru_a2,
                sB, nullptr, nullptr, nullptr, nullptr, nullptr, nullptr, 1, stream);
    launch_conv(sB, w2fi, inru_b2, CBIG, CBIG, 1, 0, /*mode*/1, nullptr,
                /*outTC=*/sA, /*resid=*/zin, nullptr, nullptr, nullptr, nullptr, nullptr, 1, stream);
    // ---- in_w: 512 -> 16, z f32 [B][T][16]
    launch_conv(sA, inwf, in_b, CBIG, CSM, 1, 0, /*mode*/3, nullptr,
                nullptr, nullptr, nullptr, /*outF32TC=*/zbuf, nullptr, nullptr, nullptr, 1, stream);

    hipMemsetAsync(zqbuf, 0, (size_t)BATCH * TLEN * CSM * 4, stream);

    // ---- 24-scale FSQ residual loop (16 channels)
    static const int SC[24] = {4096,4096,4096,4096, 1024,1024,1024,1024,
                               256,256,256,256, 16,16,16,16, 4,4,4,4, 2,2,2,2};
    size_t codeOff = (size_t)BATCH * CBIG * TLEN;
    int* codesBase = (int*)d_out;
    for (int i = 0; i < 24; ++i) {
        int scale = SC[i];
        int P = TLEN / scale;
        fsq_kernel<<<(BATCH * P + 255) / 256, 256, 0, stream>>>(
            zbuf, fsq_win, fsq_bin, fsq_wout, fsq_bout, scale, P, rbuf, codesBase + codeOff);
        upsample_snake<<<(int)((nSm + 255) / 256), 256, 0, stream>>>(rbuf, phi_a1, scale, s16a);
        launch_conv(s16a, phi1f, phi_b1, CSM, CSM, 7, 3, /*mode*/0, phi_a2,
                    s16b, nullptr, nullptr, nullptr, nullptr, nullptr, nullptr, 1, stream);
        launch_conv(s16b, phi2f, phi_b2, CSM, CSM, 1, 0, /*mode*/2, nullptr,
                    nullptr, nullptr, nullptr, nullptr, zbuf, zqbuf, rbuf, scale, stream);
        codeOff += (size_t)BATCH * P;
    }

    // ---- out_w: 16 -> 512. f32 [C][T] stashed in d_out (residual source),
    //      snake'd bf16 [T][C] into sA for out_ru conv7.
    cvt_f32_bf16<<<(int)((nSm + 255) / 256), 256, 0, stream>>>(zqbuf, zqbf, nSm);
    launch_conv(zqbf, outwf, out_b, CSM, CBIG, 1, 0, /*mode*/4, outru_a1,
                /*outTC=*/sA, nullptr, /*outCT=*/(float*)d_out, nullptr,
                nullptr, nullptr, nullptr, 1, stream);
    // ---- out_ru
    launch_conv(sA, w1fo, outru_b1, CBIG, CBIG, 7, 3, /*mode*/0, outru_a2,
                sB, nullptr, nullptr, nullptr, nullptr, nullptr, nullptr, 1, stream);
    launch_conv(sB, w2fo, outru_b2, CBIG, CBIG, 1, 0, /*mode*/1, nullptr,
                nullptr, /*resid=*/(const float*)d_out, /*outCT=*/(float*)d_out, nullptr,
                nullptr, nullptr, nullptr, 1, stream);
}
